// CausalAttention_38680475468508
// MI455X (gfx1250) — matrix operand order repaired and checked by reading, not compiled, not run on hardware
//
#include <hip/hip_runtime.h>
#include <hip/hip_bf16.h>
#include <math.h>

// ---------------------------------------------------------------------------
// Causal MHA forward for MI455X (gfx1250, wave32, WMMA bf16 16x16x32).
// Compute-bound (~69 GFLOP vs <150MB traffic, all L2-resident) -> all four
// matmul stages run on v_wmma_f32_16x16x32_bf16 with f32 accumulation.
// GEMM: 128x64 tiles, 8 waves in 4x2, 2x2 register tile per wave
//       (4 WMMAs per 4 fragment loads).  B tiles stored transposed in LDS so
//       every fragment load is consecutive ds_load_b32 pairs.
// Async global->LDS (ASYNCcnt) used for straight-copy tiles.
// ---------------------------------------------------------------------------

typedef __attribute__((ext_vector_type(16))) __bf16        v16bf;
typedef __attribute__((ext_vector_type(8)))  float         v8f;
typedef __attribute__((ext_vector_type(8)))  unsigned int  v8u;
typedef __attribute__((ext_vector_type(4)))  int           v4i_t;

#define B_   2
#define L_   2048
#define D_   1024
#define H_   16
#define HD_  64
#define MROWS (B_ * L_)     // 4096
#define NQKV  (3 * D_)      // 3072

#if __has_builtin(__builtin_amdgcn_global_load_async_to_lds_b128) && \
    __has_builtin(__builtin_amdgcn_s_wait_asynccnt)
#define HAS_ASYNC_LDS 1
#else
#define HAS_ASYNC_LDS 0
#endif

#if HAS_ASYNC_LDS
// Builtin signature (from compiler diagnostic): (v4i AS1*, v4i AS3*, Ii, Ii)
typedef __attribute__((address_space(1))) v4i_t g_v4i;
typedef __attribute__((address_space(3))) v4i_t l_v4i;
static __device__ __forceinline__ void async_copy_b128(const void* gptr, void* lptr) {
  g_v4i* g = (g_v4i*)(unsigned long long)(uintptr_t)gptr;
  l_v4i* l = (l_v4i*)(unsigned int)(uintptr_t)lptr;
  __builtin_amdgcn_global_load_async_to_lds_b128(g, l, 0, 0);
}
static __device__ __forceinline__ void async_wait0() {
  __builtin_amdgcn_s_wait_asynccnt(0);
}
#endif

static __device__ __forceinline__ unsigned short f32_to_bf16(float f) {
  unsigned int u = __builtin_bit_cast(unsigned int, f);
  u += 0x7FFFu + ((u >> 16) & 1u);          // round-to-nearest-even
  return (unsigned short)(u >> 16);
}

static __device__ __forceinline__ v8f wmma_bf16(v16bf a, v16bf b, v8f c) {
  return __builtin_amdgcn_wmma_f32_16x16x32_bf16(
      /*neg_a=*/false, a, /*neg_b=*/false, b,
      /*c_mod=*/(short)0, c, /*reuse_a=*/false, /*reuse_b=*/false);
}

// A fragment 16x32 (MxK) bf16: lane half=lane>>4, M=lane&15.
// VGPR v<4 : K = 2v + 8*half ; v>=4 : K = 16 + 2(v-4) + 8*half.
// LDS holds row-major pairs packed as uints (stride in uints).
static __device__ __forceinline__ v16bf
load_a_frag(const unsigned int* lds, int rowBase, int strideU, int kChunkU) {
  int lane = threadIdx.x & 31;
  int half = lane >> 4;
  const unsigned int* p = lds + (rowBase + (lane & 15)) * strideU + kChunkU + half * 4;
  v8u r;
  r[0] = p[0]; r[1] = p[1]; r[2] = p[2];  r[3] = p[3];
  r[4] = p[8]; r[5] = p[9]; r[6] = p[10]; r[7] = p[11];
  return __builtin_bit_cast(v16bf, r);
}

// B fragment 32x16 (KxN) from B^T stored [n][k] (pairs packed as uints).
// Lane: N=lane&15 (half group => K+=8).  VGPR v<4: K = 2v + 8*half ; v>=4: K = 16 + 2(v-4) + 8*half.
static __device__ __forceinline__ v16bf
load_b_frag_t(const unsigned int* lds, int nBase, int strideU, int kChunkU) {
  int lane = threadIdx.x & 31;
  int half = lane >> 4;
  const unsigned int* p = lds + (nBase + (lane & 15)) * strideU + kChunkU + half * 4;
  v8u r;
#pragma unroll
  for (int v = 0; v < 8; ++v) r[v] = p[v < 4 ? v : v + 4];
  return __builtin_bit_cast(v16bf, r);
}

// ---------------------------------------------------------------------------
__global__ void k_f32_to_bf16(const float* __restrict__ in,
                              unsigned short* __restrict__ out, int n) {
  int i = blockIdx.x * blockDim.x + threadIdx.x;
  int stride = gridDim.x * blockDim.x;
  for (; i < n; i += stride) out[i] = f32_to_bf16(in[i]);
}

// ---------------------------------------------------------------------------
// Tiled GEMM: C[M][N] = A[M][Kd] @ W[Kd][N] + bias.  128x64 tile, Kstep=32,
// 256 threads = 8 waves in 4(m) x 2(n); each wave owns a 2x2 grid of 16x16
// WMMA tiles (4 WMMAs per k-step, each fragment reused twice).
// mode 0: fp32 out.   mode 1: QKV epilogue -> scatter bf16 into [B][H][L][HD].
__global__ __launch_bounds__(256) void k_gemm_bf16(
    const unsigned int* __restrict__ A,   // bf16 pairs [M][Kd/2]
    const unsigned int* __restrict__ W,   // bf16 pairs [Kd][N/2]
    const float* __restrict__ bias,
    float* __restrict__ outF,
    unsigned short* __restrict__ outQ,
    unsigned short* __restrict__ outK,
    unsigned short* __restrict__ outV,
    int M, int N, int Kd, int mode) {
  __shared__ unsigned int sA[128 * 17];   // [m][k-pairs], pad 17
  __shared__ unsigned int sBt[64 * 17];   // B^T [n][k-pairs], pad 17

  int tid  = threadIdx.x;
  int lane = tid & 31, wave = tid >> 5;
  int half = lane >> 4, nl = lane & 15;
  int nBM = M >> 7;
  int bm = blockIdx.x % nBM;
  int bn = blockIdx.x / nBM;

  int mi = wave & 3;        // m-group: rows mi*32 .. mi*32+31
  int ni = wave >> 2;       // n-group: cols ni*32 .. ni*32+31

  v8f acc00 = {}, acc01 = {}, acc10 = {}, acc11 = {};
  int KdU = Kd >> 1, NU = N >> 1;

  for (int k0 = 0; k0 < Kd; k0 += 32) {
    __syncthreads();
    {  // A tile: 128 rows x 16 uints = 512 uint4; 2 transfers/thread
#pragma unroll
      for (int it = 0; it < 2; ++it) {
        int idx = tid + it * 256;
        int row = idx >> 2, cu = (idx & 3) * 4;
        const unsigned int* g = A + (size_t)(bm * 128 + row) * KdU + (k0 >> 1) + cu;
#if HAS_ASYNC_LDS
        async_copy_b128(g, &sA[row * 17 + cu]);
#else
        uint4 t = *(const uint4*)g;
        unsigned int* s = sA + row * 17 + cu;
        s[0] = t.x; s[1] = t.y; s[2] = t.z; s[3] = t.w;
#endif
      }
    }
    {  // W tile: 32 k-rows x 64 n, loaded b128 then stored transposed
      int kr = tid >> 3;
      int n0 = (tid & 7) * 8;
      const uint4* g = (const uint4*)(W + (size_t)(k0 + kr) * NU + bn * 32 + (tid & 7) * 4);
      uint4 t = g[0];
      unsigned short* sb = (unsigned short*)sBt;   // element (n,k) at n*34 + k
#pragma unroll
      for (int e = 0; e < 4; ++e) {
        unsigned int w2 = e == 0 ? t.x : e == 1 ? t.y : e == 2 ? t.z : t.w;
        int n = n0 + e * 2;
        sb[n * 34 + kr]       = (unsigned short)(w2 & 0xFFFFu);
        sb[(n + 1) * 34 + kr] = (unsigned short)(w2 >> 16);
      }
    }
    if (k0 + 32 < Kd)   // hint next W tile into cache (global_prefetch_b8)
      __builtin_prefetch(W + (size_t)(k0 + 32) * NU + bn * 32, 0, 1);
#if HAS_ASYNC_LDS
    async_wait0();
#endif
    __syncthreads();

    v16bf a0 = load_a_frag(sA, mi * 32, 17, 0);
    v16bf a1 = load_a_frag(sA, mi * 32 + 16, 17, 0);
    v16bf b0 = load_b_frag_t(sBt, ni * 32, 17, 0);
    v16bf b1 = load_b_frag_t(sBt, ni * 32 + 16, 17, 0);
    acc00 = wmma_bf16(a0, b0, acc00);
    acc01 = wmma_bf16(a0, b1, acc01);
    acc10 = wmma_bf16(a1, b0, acc10);
    acc11 = wmma_bf16(a1, b1, acc11);
  }

#pragma unroll
  for (int s = 0; s < 2; ++s) {
#pragma unroll
    for (int t = 0; t < 2; ++t) {
      v8f acc = (s == 0) ? (t == 0 ? acc00 : acc01) : (t == 0 ? acc10 : acc11);
      int n = bn * 64 + (ni * 2 + t) * 16 + nl;
      float bv = bias[n];
#pragma unroll
      for (int v = 0; v < 8; ++v) {
        int m = bm * 128 + (mi * 2 + s) * 16 + half * 8 + v;  // C layout: M=v+8*half
        float val = acc[v] + bv;
        if (mode == 0) {
          outF[(size_t)m * N + n] = val;
        } else {
          int which = n >> 10, dm = n & 1023;
          int h = dm >> 6, hd = dm & 63;
          int b = m >> 11, l = m & 2047;
          unsigned short* dst = (which == 0) ? outQ : (which == 1) ? outK : outV;
          dst[(((size_t)(b * H_ + h)) * L_ + l) * HD_ + hd] = f32_to_bf16(val);
        }
      }
    }
  }
}

// ---------------------------------------------------------------------------
// Flash attention: one block = (b, h, 64-row q tile); 128 threads = 4 waves,
// each wave owns 16 q rows.  Online softmax, all matmuls on WMMA bf16.
__global__ __launch_bounds__(128) void k_flash(
    const unsigned int* __restrict__ Q,   // bf16 pairs [B][H][L][HD/2]
    const unsigned int* __restrict__ K,
    const unsigned int* __restrict__ V,
    unsigned short* __restrict__ O) {     // bf16 [B][L][D]
  __shared__ unsigned int sQ[64 * 34];                    // [q][hd-pairs]
  __shared__ unsigned int sK[64 * 34];                    // [kpos][hd-pairs] = B^T for QK^T
  __shared__ __align__(16) unsigned short sVt[64 * 68];   // [hd][kpos]      = B^T for PV
  __shared__ unsigned int sP[4 * 16 * 34];                // per-wave P tile [16][kpos-pairs]

  int tid  = threadIdx.x;
  int lane = tid & 31, wave = tid >> 5;
  int half = lane >> 4, nl = lane & 15;

  const int nQT = L_ / 64;  // 32
  int qt = blockIdx.x % nQT;
  int h  = (blockIdx.x / nQT) % H_;
  int b  = blockIdx.x / (nQT * H_);
  size_t headBase = ((size_t)(b * H_ + h)) * L_;   // row index into [B*H][L][HD]

  {  // load Q tile
    const uint4* g4 = (const uint4*)(Q + (headBase + (size_t)qt * 64) * 32);
    for (int i = tid; i < 64 * 8; i += 128) {
      int r = i >> 3, c = (i & 7) * 4;
#if HAS_ASYNC_LDS
      async_copy_b128(g4 + i, &sQ[r * 34 + c]);
#else
      uint4 t = g4[i];
      unsigned int* s = sQ + r * 34 + c;
      s[0] = t.x; s[1] = t.y; s[2] = t.z; s[3] = t.w;
#endif
    }
  }
#if HAS_ASYNC_LDS
  async_wait0();
#endif
  __syncthreads();

  v16bf qa0 = load_a_frag(sQ, wave * 16, 34, 0);
  v16bf qa1 = load_a_frag(sQ, wave * 16, 34, 16);

  float mrun[8], lrun[8];
  v8f oacc[4];
#pragma unroll
  for (int v = 0; v < 8; ++v) { mrun[v] = -3.0e38f; lrun[v] = 0.0f; }
#pragma unroll
  for (int jj = 0; jj < 4; ++jj) { v8f z = {}; oacc[jj] = z; }

  for (int kt = 0; kt <= qt; ++kt) {
    __syncthreads();
    {  // K tile row-major
      const uint4* g4 = (const uint4*)(K + (headBase + (size_t)kt * 64) * 32);
      for (int i = tid; i < 64 * 8; i += 128) {
        int r = i >> 3, c = (i & 7) * 4;
#if HAS_ASYNC_LDS
        async_copy_b128(g4 + i, &sK[r * 34 + c]);
#else
        uint4 t = g4[i];
        unsigned int* s = sK + r * 34 + c;
        s[0] = t.x; s[1] = t.y; s[2] = t.z; s[3] = t.w;
#endif
      }
      // V tile transposed into [hd][kpos]
      const unsigned int* gv = V + (headBase + (size_t)kt * 64) * 32;
      for (int i = tid; i < 64 * 32; i += 128) {
        int kp = i >> 5, cu = i & 31;
        unsigned int w2 = gv[kp * 32 + cu];
        int hd = cu * 2;
        sVt[hd * 68 + kp]       = (unsigned short)(w2 & 0xFFFFu);
        sVt[(hd + 1) * 68 + kp] = (unsigned short)(w2 >> 16);
      }
    }
#if HAS_ASYNC_LDS
    async_wait0();
#endif
    __syncthreads();

    // ---- S = Q @ K^T  (4 n-subtiles x 2 k-steps = 8 WMMAs) ----
    v8f s[4];
#pragma unroll
    for (int j = 0; j < 4; ++j) {
      v8f z = {};
      z = wmma_bf16(qa0, load_b_frag_t(sK, j * 16, 34, 0),  z);
      z = wmma_bf16(qa1, load_b_frag_t(sK, j * 16, 34, 16), z);
      s[j] = z;
    }

    // ---- scale + causal mask ----
    bool diag = (kt == qt);
#pragma unroll
    for (int j = 0; j < 4; ++j)
#pragma unroll
      for (int v = 0; v < 8; ++v) {
        float x = s[j][v] * 0.125f;            // 1/sqrt(64)
        if (diag) {
          int n_g = j * 16 + nl;
          int m_g = wave * 16 + half * 8 + v;
          if (n_g > m_g) x = -3.0e38f;
        }
        s[j][v] = x;
      }

    // ---- online softmax (row reductions across 16-lane half-groups) ----
    float mnew[8], scaleOld[8];
#pragma unroll
    for (int v = 0; v < 8; ++v) {
      float pmv = fmaxf(fmaxf(s[0][v], s[1][v]), fmaxf(s[2][v], s[3][v]));
#pragma unroll
      for (int off = 8; off >= 1; off >>= 1)
        pmv = fmaxf(pmv, __shfl_xor(pmv, off, 32));
      mnew[v] = fmaxf(mrun[v], pmv);
      scaleOld[v] = __expf(mrun[v] - mnew[v]);
    }

    unsigned short* pw = (unsigned short*)&sP[wave * 16 * 34];
#pragma unroll
    for (int v = 0; v < 8; ++v) {
      float rs = 0.0f;
#pragma unroll
      for (int j = 0; j < 4; ++j) {
        float p = __expf(s[j][v] - mnew[v]);
        rs += p;
        pw[(half * 8 + v) * 68 + j * 16 + nl] = f32_to_bf16(p);  // C-layout -> LDS
      }
#pragma unroll
      for (int off = 8; off >= 1; off >>= 1)
        rs += __shfl_xor(rs, off, 32);
      lrun[v] = lrun[v] * scaleOld[v] + rs;
      mrun[v] = mnew[v];
    }

    // rescale accumulators (row v+8*half matches C layout)
#pragma unroll
    for (int jj = 0; jj < 4; ++jj)
#pragma unroll
      for (int v = 0; v < 8; ++v)
        oacc[jj][v] *= scaleOld[v];

    // ---- O += P @ V  (reload P as A-fragments; 8 WMMAs) ----
    const unsigned int* pr = &sP[wave * 16 * 34];
    v16bf pa0 = load_a_frag(pr, 0, 34, 0);
    v16bf pa1 = load_a_frag(pr, 0, 34, 16);
    const unsigned int* vt = (const unsigned int*)sVt;
#pragma unroll
    for (int jj = 0; jj < 4; ++jj) {
      oacc[jj] = wmma_bf16(pa0, load_b_frag_t(vt, jj * 16, 34, 0),  oacc[jj]);
      oacc[jj] = wmma_bf16(pa1, load_b_frag_t(vt, jj * 16, 34, 16), oacc[jj]);
    }
  }

  // ---- epilogue: O /= l ; write bf16 [B][L][D] at column h*64+hd ----
#pragma unroll
  for (int v = 0; v < 8; ++v) {
    float inv = 1.0f / lrun[v];
    int l = qt * 64 + wave * 16 + half * 8 + v;
    size_t rowBase = ((size_t)b * L_ + l) * D_ + h * 64;
#pragma unroll
    for (int jj = 0; jj < 4; ++jj)
      O[rowBase + jj * 16 + nl] = f32_to_bf16(oacc[jj][v] * inv);
  }
}

// ---------------------------------------------------------------------------
extern "C" void kernel_launch(void* const* d_in, const int* in_sizes, int n_in,
                              void* d_out, int out_size, void* d_ws, size_t ws_size,
                              hipStream_t stream) {
  const float* x     = (const float*)d_in[0];
  const float* w_qkv = (const float*)d_in[1];
  const float* b_qkv = (const float*)d_in[2];
  const float* w_out = (const float*)d_in[3];
  const float* b_out = (const float*)d_in[4];
  float* out = (float*)d_out;

  char* ws = (char*)d_ws;
  const size_t MB = 1024 * 1024;
  unsigned short* xbf   = (unsigned short*)(ws + 0 * MB);   // 8 MB  [B*L][D]
  unsigned short* wqkvb = (unsigned short*)(ws + 8 * MB);   // 6 MB  [D][3D]
  unsigned short* woutb = (unsigned short*)(ws + 14 * MB);  // 2 MB  [D][D]
  unsigned short* qb    = (unsigned short*)(ws + 16 * MB);  // 8 MB  [B][H][L][HD]
  unsigned short* kb    = (unsigned short*)(ws + 24 * MB);  // 8 MB
  unsigned short* vb    = (unsigned short*)(ws + 32 * MB);  // 8 MB
  unsigned short* ab    = (unsigned short*)(ws + 40 * MB);  // 8 MB  [B*L][D]

  k_f32_to_bf16<<<1024, 256, 0, stream>>>(x, xbf, MROWS * D_);
  k_f32_to_bf16<<<1024, 256, 0, stream>>>(w_qkv, wqkvb, D_ * NQKV);
  k_f32_to_bf16<<<1024, 256, 0, stream>>>(w_out, woutb, D_ * D_);

  // QKV projection: [4096 x 3072] = xbf @ wqkvb  (+bias), scatter to Q/K/V
  k_gemm_bf16<<<(MROWS / 128) * (NQKV / 64), 256, 0, stream>>>(
      (const unsigned int*)xbf, (const unsigned int*)wqkvb, b_qkv,
      nullptr, qb, kb, vb, MROWS, NQKV, D_, 1);

  // Causal flash attention
  k_flash<<<B_ * H_ * (L_ / 64), 128, 0, stream>>>(
      (const unsigned int*)qb, (const unsigned int*)kb, (const unsigned int*)vb, ab);

  // Output projection: [4096 x 1024] = ab @ woutb (+bias) -> fp32 d_out
  k_gemm_bf16<<<(MROWS / 128) * (D_ / 64), 256, 0, stream>>>(
      (const unsigned int*)ab, (const unsigned int*)woutb, b_out,
      out, nullptr, nullptr, nullptr, MROWS, D_, D_, 0);
}
